// PNANet_81595788689986
// MI455X (gfx1250) — matrix operand order repaired and checked by reading, not compiled, not run on hardware
//
#include <hip/hip_runtime.h>
#include <math.h>

#define N_NODES 50000
#define N_EDGES 800000
#define FDIM 64
#define EDIM 16
#define NLAYERS 2
#define NF (N_NODES * FDIM)
#define NTILES ((N_NODES + 15) / 16) /* 3125 */
#define EPS_BN 1e-5f

typedef __attribute__((ext_vector_type(16))) __bf16    v16bf;
typedef __attribute__((ext_vector_type(8)))  float     v8f;
typedef __attribute__((ext_vector_type(8)))  unsigned  v8u;

// ---------- helpers ----------

// native packed f32 -> bf16 conversion (RNE in hardware), 1 VALU op per pair
__device__ __forceinline__ unsigned pk2(float lo, float hi) {
  unsigned r;
  asm("v_cvt_pk_bf16_f32 %0, %1, %2" : "=v"(r) : "v"(lo), "v"(hi));
  return r;
}

// A-fragment (bf16 16x32): elements 0-7 = K run at half*8, elements 8-15 = K run at 16+half*8
__device__ __forceinline__ v16bf afrag(float4 q0, float4 q1, float4 q2, float4 q3) {
  v8u w;
  w[0] = pk2(q0.x, q0.y); w[1] = pk2(q0.z, q0.w);
  w[2] = pk2(q1.x, q1.y); w[3] = pk2(q1.z, q1.w);
  w[4] = pk2(q2.x, q2.y); w[5] = pk2(q2.z, q2.w);
  w[6] = pk2(q3.x, q3.y); w[7] = pk2(q3.z, q3.w);
  return __builtin_bit_cast(v16bf, w);
}

// B-fragment: pre-packed 8 dwords per lane
__device__ __forceinline__ v16bf bfrag(const unsigned* __restrict__ p) {
  uint4 a = *(const uint4*)p;
  uint4 b = *(const uint4*)(p + 4);
  v8u w;
  w[0] = a.x; w[1] = a.y; w[2] = a.z; w[3] = a.w;
  w[4] = b.x; w[5] = b.y; w[6] = b.z; w[7] = b.w;
  return __builtin_bit_cast(v16bf, w);
}

__device__ __forceinline__ v8f wmma_bf16(v16bf a, v16bf b, v8f c) {
  return __builtin_amdgcn_wmma_f32_16x16x32_bf16(false, a, false, b, (short)0, c, false, false);
}

// order-preserving float<->uint map for integer min/max atomics
__device__ __forceinline__ unsigned ford(float f) {
  unsigned u = __float_as_uint(f);
  return (u & 0x80000000u) ? ~u : (u | 0x80000000u);
}
__device__ __forceinline__ float funord(unsigned u) {
  return __uint_as_float((u & 0x80000000u) ? (u ^ 0x80000000u) : ~u);
}

// native no-return global atomics (CDNA5)
__device__ __forceinline__ void gadd_f32(float* p, float v) {
  asm volatile("global_atomic_add_f32 %0, %1, off" :: "v"(p), "v"(v) : "memory");
}
__device__ __forceinline__ void gmin_u32(unsigned* p, unsigned v) {
  asm volatile("global_atomic_min_u32 %0, %1, off" :: "v"(p), "v"(v) : "memory");
}
__device__ __forceinline__ void gmax_u32(unsigned* p, unsigned v) {
  asm volatile("global_atomic_max_u32 %0, %1, off" :: "v"(p), "v"(v) : "memory");
}

// ---------- small utility kernels ----------

__global__ void k_fill(unsigned* __restrict__ p, unsigned val, int n) {
  int id = blockIdx.x * 256 + threadIdx.x;
  if (id < n) p[id] = val;
}

__global__ void k_count(const int* __restrict__ dst, unsigned* __restrict__ icnt) {
  int id = blockIdx.x * 256 + threadIdx.x;
  if (id < N_EDGES) atomicAdd(&icnt[dst[id]], 1u);
}

__global__ void k_bincount(const unsigned* __restrict__ icnt, unsigned* __restrict__ bc) {
  int id = blockIdx.x * 256 + threadIdx.x;
  if (id < N_NODES) {
    unsigned c = icnt[id];
    if (c > (unsigned)(N_NODES - 1)) c = N_NODES - 1;
    atomicAdd(&bc[c], 1u);
  }
}

__global__ void k_sumlog(const unsigned* __restrict__ bc, float* __restrict__ scal) {
  __shared__ float sm[256];
  int id = blockIdx.x * 256 + threadIdx.x;
  float v = (id < N_NODES) ? logf((float)bc[id] + 1.0f) : 0.0f;
  sm[threadIdx.x] = v;
  __syncthreads();
  for (int s = 128; s > 0; s >>= 1) {
    if ((int)threadIdx.x < s) sm[threadIdx.x] += sm[threadIdx.x + s];
    __syncthreads();
  }
  if (threadIdx.x == 0) gadd_f32(&scal[0], sm[0]);
}

__global__ void k_amp(const unsigned* __restrict__ icnt, const float* __restrict__ scal,
                      float* __restrict__ amp) {
  int id = blockIdx.x * 256 + threadIdx.x;
  if (id < N_NODES) {
    unsigned c = icnt[id]; if (c < 1u) c = 1u;
    // amp = log(cnt+1) / (sumlog/N)
    amp[id] = logf((float)c + 1.0f) * ((float)N_NODES / scal[0]);
  }
}

// W2' = We @ Wpre[64:128]  (16x64),  b' = be @ Wpre[64:128] + bpre
__global__ void k_fuse(const float* __restrict__ We_l, const float* __restrict__ Wpre_l,
                       const float* __restrict__ be_l, const float* __restrict__ bpre_l,
                       float* __restrict__ W2raw, float* __restrict__ bfused) {
  int id = blockIdx.x * 256 + threadIdx.x;
  const float* wb = Wpre_l + 64 * 64;  // bottom 64 rows
  if (id < 1024) {
    int k = id >> 6, n = id & 63;
    float acc = 0.0f;
    for (int j = 0; j < 64; ++j) acc += We_l[k * 64 + j] * wb[j * 64 + n];
    W2raw[id] = acc;
  } else if (id < 1088) {
    int n = id - 1024;
    float acc = bpre_l[n];
    for (int j = 0; j < 64; ++j) acc += be_l[j] * wb[j * 64 + n];
    bfused[n] = acc;
  }
}

// pack f32 weight [K,64] row-major into bf16 B-fragments:
// dst word index = ((kt*4 + nt)*32 + lane)*8 + v ; K=kt*32+half*8+2v (v<4), kt*32+16+half*8+2(v-4) (v>=4)
__global__ void k_pack(const float* __restrict__ W, int K, unsigned* __restrict__ dst, int total) {
  int id = blockIdx.x * 256 + threadIdx.x;
  if (id >= total) return;
  int v    = id & 7;
  int lane = (id >> 3) & 31;
  int nt   = (id >> 8) & 3;
  int kt   = id >> 10;
  int n  = nt * 16 + (lane & 15);
  int kb = kt * 32 + ((lane >> 4) << 3) + 2 * v + ((v >> 2) << 3);
  float f0 = (kb     < K) ? W[(size_t)kb * 64 + n]       : 0.0f;
  float f1 = (kb + 1 < K) ? W[(size_t)(kb + 1) * 64 + n] : 0.0f;
  dst[id] = pk2(f0, f1);
}

__global__ void k_finalize(float* __restrict__ sumb, unsigned* __restrict__ minb,
                           unsigned* __restrict__ maxb, const unsigned* __restrict__ icnt) {
  int id = blockIdx.x * 256 + threadIdx.x;
  if (id >= NF) return;
  unsigned c = icnt[id >> 6];
  float inv = 1.0f / (float)(c ? c : 1u);
  sumb[id] *= inv;                      // sum -> mean
  float mn = c ? funord(minb[id]) : 0.0f;
  float mx = c ? funord(maxb[id]) : 0.0f;
  ((float*)minb)[id] = mn;
  ((float*)maxb)[id] = mx;
}

__global__ void k_bn(const float* __restrict__ Y, const float* __restrict__ bnsum,
                     const float* __restrict__ bnss, const float* __restrict__ gamma,
                     const float* __restrict__ beta, float* __restrict__ out) {
  int id = blockIdx.x * 256 + threadIdx.x;
  if (id >= NF) return;
  int f = id & 63;
  const float invN = 1.0f / (float)N_NODES;
  float mu  = bnsum[f] * invN;
  float var = bnss[f] * invN - mu * mu;
  float v = (Y[id] - mu) * rsqrtf(var + EPS_BN) * gamma[f] + beta[f];
  out[id] = fmaxf(v, 0.0f);
}

// ---------- edge message kernel: h = x[src]@W1 + [EA;0]@W2' + b' , fused scatter ----------

__global__ __launch_bounds__(256) void k_edge(
    const float* __restrict__ X, const int* __restrict__ src, const int* __restrict__ dst,
    const float* __restrict__ EA, const unsigned* __restrict__ pW1,
    const unsigned* __restrict__ pW2, const float* __restrict__ bfused,
    float* __restrict__ sumb, unsigned* __restrict__ minb, unsigned* __restrict__ maxb) {
  const int lane = threadIdx.x & 31;
  const int wave = threadIdx.x >> 5;
  const int m  = lane & 15;
  const int hi = lane >> 4;
  const long e0 = ((long)blockIdx.x * 8 + wave) * 16;
  const long e  = e0 + m;
  const int s = src[e];
  const int d = dst[e];

  v8f c[4];
#pragma unroll
  for (int nt = 0; nt < 4; ++nt) {
    float b = bfused[nt * 16 + m];
#pragma unroll
    for (int r = 0; r < 8; ++r) c[nt][r] = b;
  }

  // x[src] @ W1 : K=64, 2 k-tiles
  const float* xr = X + (size_t)s * FDIM;
#pragma unroll
  for (int kt = 0; kt < 2; ++kt) {
    const float* p = xr + kt * 32 + hi * 8;
    float4 q0 = *(const float4*)(p);
    float4 q1 = *(const float4*)(p + 4);
    float4 q2 = *(const float4*)(p + 16);
    float4 q3 = *(const float4*)(p + 20);
    v16bf a = afrag(q0, q1, q2, q3);
#pragma unroll
    for (int nt = 0; nt < 4; ++nt) {
      v16bf b = bfrag(pW1 + ((size_t)((kt * 4 + nt) * 32 + lane)) * 8);
      c[nt] = wmma_bf16(a, b, c[nt]);
    }
  }

  // edge_attr @ W2' : K=16 zero-padded to 32
  {
    const float* p = EA + (size_t)e * EDIM + hi * 8;
    float4 q0 = *(const float4*)(p);
    float4 q1 = *(const float4*)(p + 4);
    float4 z  = make_float4(0.f, 0.f, 0.f, 0.f);
    v16bf a = afrag(q0, q1, z, z);
#pragma unroll
    for (int nt = 0; nt < 4; ++nt) {
      v16bf b = bfrag(pW2 + ((size_t)(nt * 32 + lane)) * 8);
      c[nt] = wmma_bf16(a, b, c[nt]);
    }
  }

  // fused scatter: mean-sum + min + max per dst node
#pragma unroll
  for (int nt = 0; nt < 4; ++nt) {
    int f = nt * 16 + m;
#pragma unroll
    for (int r = 0; r < 8; ++r) {
      float v  = c[nt][r];
      int   dd = __shfl(d, hi * 8 + r, 32);  // dst of row (hi*8+r) of this tile
      size_t idx = (size_t)dd * FDIM + f;
      gadd_f32(&sumb[idx], v);
      unsigned o = ford(v);
      gmin_u32(&minb[idx], o);
      gmax_u32(&maxb[idx], o);
    }
  }
}

// ---------- node kernel: y = (cat(x, agg6) @ Wpost + bpost) @ Wlin + blin, BN partials ----------

__global__ __launch_bounds__(256) void k_node(
    const float* __restrict__ X, const float* __restrict__ meanb,
    const float* __restrict__ mnb, const float* __restrict__ mxb,
    const float* __restrict__ amp,
    const unsigned* __restrict__ pWpost, const float* __restrict__ bpost,
    const unsigned* __restrict__ pWlin,  const float* __restrict__ blin,
    float* __restrict__ Y, float* __restrict__ bnsum, float* __restrict__ bnss) {
  __shared__ float lds[8 * 16 * FDIM];  // 32 KB: one 16x64 f32 tile per wave
  const int lane = threadIdx.x & 31;
  const int wave = threadIdx.x >> 5;
  const int m  = lane & 15;
  const int hi = lane >> 4;
  int tile = blockIdx.x * 8 + wave;
  const bool valid = (tile < NTILES);
  if (!valid) tile = NTILES - 1;
  const int node = tile * 16 + m;
  const float av = amp[node];

  const float* bases[7];
  bases[0] = X     + (size_t)node * FDIM;
  bases[1] = meanb + (size_t)node * FDIM;
  bases[2] = mnb   + (size_t)node * FDIM;
  bases[3] = mxb   + (size_t)node * FDIM;
  bases[4] = bases[1];
  bases[5] = bases[2];
  bases[6] = bases[3];

  v8f c[4];
#pragma unroll
  for (int nt = 0; nt < 4; ++nt) {
    float b = bpost[nt * 16 + m];
#pragma unroll
    for (int r = 0; r < 8; ++r) c[nt][r] = b;
  }

  // GEMM 1: K = 7*64 = 448 -> 14 k-tiles of 32; seg = kt/2 selects source buffer
#pragma unroll
  for (int kt = 0; kt < 14; ++kt) {
    const int   seg = kt >> 1;
    const float sc  = (seg >= 4) ? av : 1.0f;
    const float* p = bases[seg] + ((kt & 1) << 5) + (hi << 3);
    float4 q0 = *(const float4*)(p);
    float4 q1 = *(const float4*)(p + 4);
    float4 q2 = *(const float4*)(p + 16);
    float4 q3 = *(const float4*)(p + 20);
    q0.x *= sc; q0.y *= sc; q0.z *= sc; q0.w *= sc;
    q1.x *= sc; q1.y *= sc; q1.z *= sc; q1.w *= sc;
    q2.x *= sc; q2.y *= sc; q2.z *= sc; q2.w *= sc;
    q3.x *= sc; q3.y *= sc; q3.z *= sc; q3.w *= sc;
    v16bf a = afrag(q0, q1, q2, q3);
#pragma unroll
    for (int nt = 0; nt < 4; ++nt) {
      v16bf b = bfrag(pWpost + ((size_t)((kt * 4 + nt) * 32 + lane)) * 8);
      c[nt] = wmma_bf16(a, b, c[nt]);
    }
  }

  // stage y1 tile in LDS (C layout -> row major) for A-fragment re-read
  float* my = lds + wave * (16 * FDIM);
#pragma unroll
  for (int nt = 0; nt < 4; ++nt)
#pragma unroll
    for (int r = 0; r < 8; ++r)
      my[(hi * 8 + r) * FDIM + nt * 16 + m] = c[nt][r];
  asm volatile("s_wait_dscnt 0" ::: "memory");

  // GEMM 2: y = y1 @ Wlin + blin, K=64
  v8f c2[4];
#pragma unroll
  for (int nt = 0; nt < 4; ++nt) {
    float b = blin[nt * 16 + m];
#pragma unroll
    for (int r = 0; r < 8; ++r) c2[nt][r] = b;
  }
#pragma unroll
  for (int kt = 0; kt < 2; ++kt) {
    const float* p = my + m * FDIM + kt * 32 + hi * 8;
    float4 q0 = *(const float4*)(p);
    float4 q1 = *(const float4*)(p + 4);
    float4 q2 = *(const float4*)(p + 16);
    float4 q3 = *(const float4*)(p + 20);
    v16bf a = afrag(q0, q1, q2, q3);
#pragma unroll
    for (int nt = 0; nt < 4; ++nt) {
      v16bf b = bfrag(pWlin + ((size_t)((kt * 4 + nt) * 32 + lane)) * 8);
      c2[nt] = wmma_bf16(a, b, c2[nt]);
    }
  }

  // store y + accumulate BN partial sums (per feature column)
  if (valid) {
#pragma unroll
    for (int nt = 0; nt < 4; ++nt) {
      int f = nt * 16 + m;
      float s1 = 0.0f, s2 = 0.0f;
#pragma unroll
      for (int r = 0; r < 8; ++r) {
        int row = tile * 16 + hi * 8 + r;
        float v = c2[nt][r];
        if (row < N_NODES) {
          Y[(size_t)row * FDIM + f] = v;
          s1 += v; s2 += v * v;
        }
      }
      gadd_f32(&bnsum[f], s1);
      gadd_f32(&bnss[f], s2);
    }
  }
}

// ---------- host launcher ----------

extern "C" void kernel_launch(void* const* d_in, const int* in_sizes, int n_in,
                              void* d_out, int out_size, void* d_ws, size_t ws_size,
                              hipStream_t stream) {
  (void)in_sizes; (void)n_in; (void)out_size; (void)ws_size;
  const float* x     = (const float*)d_in[0];
  const int*   ei    = (const int*)d_in[1];
  const float* ea    = (const float*)d_in[2];
  const float* We    = (const float*)d_in[3];
  const float* be    = (const float*)d_in[4];
  const float* Wpre  = (const float*)d_in[5];
  const float* bpre  = (const float*)d_in[6];
  const float* Wpost = (const float*)d_in[7];
  const float* bpost = (const float*)d_in[8];
  const float* Wlin  = (const float*)d_in[9];
  const float* blin  = (const float*)d_in[10];
  const float* gamma = (const float*)d_in[11];
  const float* beta  = (const float*)d_in[12];
  const int* srcI = ei;
  const int* dstI = ei + N_EDGES;

  // workspace carve-out (256B aligned)
  size_t off = 0;
  char* ws = (char*)d_ws;
  auto alloc = [&](size_t bytes) -> void* {
    void* p = ws + off;
    off = (off + bytes + 255) & ~(size_t)255;
    return p;
  };
  float*    sumb = (float*)alloc((size_t)NF * 4);
  unsigned* minb = (unsigned*)alloc((size_t)NF * 4);
  unsigned* maxb = (unsigned*)alloc((size_t)NF * 4);
  float*    Ybuf = (float*)alloc((size_t)NF * 4);
  float*    X1   = (float*)alloc((size_t)NF * 4);
  unsigned* icnt = (unsigned*)alloc((size_t)N_NODES * 4);
  unsigned* bc   = (unsigned*)alloc((size_t)N_NODES * 4);
  float*    amp  = (float*)alloc((size_t)N_NODES * 4);
  float*    scal = (float*)alloc(256);
  float*    bnsum = (float*)alloc(64 * 4);
  float*    bnss  = (float*)alloc(64 * 4);
  unsigned* pW1[NLAYERS]; unsigned* pW2[NLAYERS];
  unsigned* pWpostB[NLAYERS]; unsigned* pWlinB[NLAYERS];
  float* W2raw[NLAYERS]; float* bfusedB[NLAYERS];
  for (int l = 0; l < NLAYERS; ++l) {
    pW1[l]     = (unsigned*)alloc(2048 * 4);
    pW2[l]     = (unsigned*)alloc(1024 * 4);
    pWpostB[l] = (unsigned*)alloc(14336 * 4);
    pWlinB[l]  = (unsigned*)alloc(2048 * 4);
    W2raw[l]   = (float*)alloc(1024 * 4);
    bfusedB[l] = (float*)alloc(64 * 4);
  }

  const dim3 B(256);
  const int gN  = (N_NODES + 255) / 256;
  const int gNF = (NF + 255) / 256;

  // degree / amplification stats: depend only on dst -> computed once
  k_fill<<<gN, B, 0, stream>>>(icnt, 0u, N_NODES);
  k_fill<<<gN, B, 0, stream>>>(bc, 0u, N_NODES);
  k_fill<<<1, B, 0, stream>>>((unsigned*)scal, 0u, 4);
  k_count<<<(N_EDGES + 255) / 256, B, 0, stream>>>(dstI, icnt);
  k_bincount<<<gN, B, 0, stream>>>(icnt, bc);
  k_sumlog<<<gN, B, 0, stream>>>(bc, scal);
  k_amp<<<gN, B, 0, stream>>>(icnt, scal, amp);

  const float* Xcur = x;
  for (int l = 0; l < NLAYERS; ++l) {
    const float* We_l    = We    + (size_t)l * EDIM * FDIM;
    const float* be_l    = be    + (size_t)l * FDIM;
    const float* Wpre_l  = Wpre  + (size_t)l * 2 * FDIM * FDIM;
    const float* bpre_l  = bpre  + (size_t)l * FDIM;
    const float* Wpost_l = Wpost + (size_t)l * 7 * FDIM * FDIM;
    const float* bpost_l = bpost + (size_t)l * FDIM;
    const float* Wlin_l  = Wlin  + (size_t)l * FDIM * FDIM;
    const float* blin_l  = blin  + (size_t)l * FDIM;

    // weight fusion + fragment packing
    k_fuse<<<5, B, 0, stream>>>(We_l, Wpre_l, be_l, bpre_l, W2raw[l], bfusedB[l]);
    k_pack<<<8,  B, 0, stream>>>(Wpre_l,   64,  pW1[l],     2048);   // top 64 rows of Wpre
    k_pack<<<4,  B, 0, stream>>>(W2raw[l], 16,  pW2[l],     1024);   // padded to K=32
    k_pack<<<56, B, 0, stream>>>(Wpost_l,  448, pWpostB[l], 14336);
    k_pack<<<8,  B, 0, stream>>>(Wlin_l,   64,  pWlinB[l],  2048);

    // per-layer init
    k_fill<<<gNF, B, 0, stream>>>((unsigned*)sumb, 0u, NF);
    k_fill<<<gNF, B, 0, stream>>>(minb, 0xFF800000u, NF);  // ford(+inf)
    k_fill<<<gNF, B, 0, stream>>>(maxb, 0x007FFFFFu, NF);  // ford(-inf)
    k_fill<<<1, B, 0, stream>>>((unsigned*)bnsum, 0u, 64);
    k_fill<<<1, B, 0, stream>>>((unsigned*)bnss, 0u, 64);

    // edge messages + scatter reduce (16 edges per wave, 8 waves per block)
    k_edge<<<N_EDGES / 128, B, 0, stream>>>(Xcur, srcI, dstI, ea, pW1[l], pW2[l],
                                            bfusedB[l], sumb, minb, maxb);
    k_finalize<<<gNF, B, 0, stream>>>(sumb, minb, maxb, icnt);

    // node MLP (two chained WMMA GEMMs) + BN partials
    k_node<<<(NTILES + 7) / 8, B, 0, stream>>>(Xcur, sumb, (const float*)minb,
                                               (const float*)maxb, amp,
                                               pWpostB[l], bpost_l, pWlinB[l], blin_l,
                                               Ybuf, bnsum, bnss);

    float* Xnext = (l == NLAYERS - 1) ? (float*)d_out : X1;
    k_bn<<<gNF, B, 0, stream>>>(Ybuf, bnsum, bnss, gamma + (size_t)l * FDIM,
                                beta + (size_t)l * FDIM, Xnext);
    Xcur = Xnext;
  }
}
